// MultiHeadSelfAttention_46935402610795
// MI455X (gfx1250) — matrix operand order repaired and checked by reading, not compiled, not run on hardware
//
#include <hip/hip_runtime.h>

#define DM   512
#define SEQ  2048
#define BATCH 2
#define NH   8
#define DH   64

typedef __attribute__((ext_vector_type(16))) _Float16 v16h;
typedef __attribute__((ext_vector_type(8)))  _Float16 v8h;
typedef __attribute__((ext_vector_type(4)))  _Float16 v4h;
typedef __attribute__((ext_vector_type(8)))  float    v8f;
typedef __attribute__((ext_vector_type(4)))  float    v4f;
typedef __attribute__((ext_vector_type(4)))  unsigned int v4u;
typedef __attribute__((ext_vector_type(8)))  int      v8i;
typedef __attribute__((ext_vector_type(4)))  int      v4i;

// ---- WMMA helpers: layouts per CDNA5 ISA 7.12.2 (wave32) -------------------

static __device__ __forceinline__ v8f wmma_f16(v16h a, v16h b, v8f c) {
  // D = A(16x32 f16) * B(32x16 f16) + C(16x16 f32)
  return __builtin_amdgcn_wmma_f32_16x16x32_f16(false, a, false, b, (short)0, c,
                                                false, false);
}

// A tile (16x32 f16) from row-major f16 [rows, ld]; chunk base kc.
// lane<16: row=lane, K = kc+0..7 (VGPR0-3) and kc+16..23 (VGPR4-7)
// lane>=16: row=lane-16, K = kc+8..15 and kc+24..31
static __device__ __forceinline__ v16h load_a_f16(const _Float16* base, int ld, int kc) {
  int lane = threadIdx.x & 31;
  int row  = lane & 15;
  int hi   = (lane >> 4) & 1;
  const _Float16* p = base + (size_t)row * ld + kc + hi * 8;
  v8h lo = *(const v8h*)(p);
  v8h hh = *(const v8h*)(p + 16);
  v16h r;
#pragma unroll
  for (int i = 0; i < 8; ++i) { r[i] = lo[i]; r[8 + i] = hh[i]; }
  return r;
}

// A tile from row-major fp32 (global or LDS), converted to f16 on the fly.
static __device__ __forceinline__ v16h load_a_f32(const float* base, int ld, int kc) {
  int lane = threadIdx.x & 31;
  int row  = lane & 15;
  int hi   = (lane >> 4) & 1;
  const float* p = base + (size_t)row * ld + kc + hi * 8;
  v4f x0 = *(const v4f*)(p);
  v4f x1 = *(const v4f*)(p + 4);
  v4f x2 = *(const v4f*)(p + 16);
  v4f x3 = *(const v4f*)(p + 20);
  v16h r;
#pragma unroll
  for (int i = 0; i < 4; ++i) {
    r[i]      = (_Float16)x0[i];
    r[4 + i]  = (_Float16)x1[i];
    r[8 + i]  = (_Float16)x2[i];
    r[12 + i] = (_Float16)x3[i];
  }
  return r;
}

// B tile (32x16 f16) from K-contiguous storage Bt[n][k] (ld = stride between n).
// lane<16: N=lane, K = kc+0..7,kc+16..23 ; lane>=16: N=lane-16, K = kc+8..15,kc+24..31
static __device__ __forceinline__ v16h load_b_f16(const _Float16* bt, int ld, int kc) {
  int lane = threadIdx.x & 31;
  int n    = lane & 15;
  int hi   = (lane >> 4) & 1;
  const _Float16* p = bt + (size_t)n * ld + kc + hi * 8;
  v8h lo = *(const v8h*)(p);
  v8h hh = *(const v8h*)(p + 16);
  v16h r;
#pragma unroll
  for (int i = 0; i < 8; ++i) { r[i] = lo[i]; r[8 + i] = hh[i]; }
  return r;
}

// ---- Tensor Data Mover: contiguous 2-byte-element 1D tile, global -> LDS ---
// D# built per CDNA5 ISA ch.8 (group0: count/lds_addr/global_addr/type,
// group1: data_size=2B, tensor_dim0=tile_dim0=nelem, dim1=1, stride0=nelem).
// Issue from ONE wave; completion via TENSORcnt, then workgroup barrier.
static __device__ __forceinline__ void tdm_copy_to_lds(const void* gsrc,
                                                       unsigned ldsByteOff,
                                                       unsigned nbytes) {
  unsigned nelem = nbytes >> 1;                       // 2-byte elements (<=65535)
  unsigned long long ga = (unsigned long long)(uintptr_t)gsrc;
  v4u g0;
  g0[0] = 1u;                                        // count=1, gather off
  g0[1] = ldsByteOff;                                // lds_addr  (bits 63:32)
  g0[2] = (unsigned)(ga & 0xffffffffu);              // global_addr[31:0]
  g0[3] = (unsigned)((ga >> 32) & 0x01ffffffu) | 0x80000000u; // addr[56:32]|type=2
  v8i g1;
  g1[0] = 0x00010000;                                // data_size=1 (2B), mask=0
  g1[1] = (int)((nelem & 0xffffu) << 16);            // tensor_dim0[15:0]
  g1[2] = (int)(((nelem >> 16) & 0xffffu) | (1u << 16)); // dim0[31:16] | tensor_dim1=1
  g1[3] = (int)((nelem & 0xffffu) << 16);            // tile_dim0
  g1[4] = 1;                                         // tile_dim1=1
  g1[5] = (int)nelem;                                // tensor_dim0_stride[31:0]
  g1[6] = 0;
  g1[7] = 0;
  v4i z4 = {0, 0, 0, 0};
#if __clang_major__ >= 23
  v8i z8 = {0, 0, 0, 0, 0, 0, 0, 0};
  __builtin_amdgcn_tensor_load_to_lds(g0, g1, z4, z4, z8, 0);
#else
  __builtin_amdgcn_tensor_load_to_lds(g0, g1, z4, z4, 0);
#endif
}

// ---- Kernel 0: convert + transpose weights to f16 [n][k] layout ------------

__global__ __launch_bounds__(256)
void prep_weights(const float* __restrict__ Wq, const float* __restrict__ Wk,
                  const float* __restrict__ Wv, const float* __restrict__ Wo,
                  _Float16* __restrict__ wtq, _Float16* __restrict__ wtk,
                  _Float16* __restrict__ wtv, _Float16* __restrict__ wto) {
  int idx = blockIdx.x * blockDim.x + threadIdx.x;
  if (idx >= 4 * DM * DM) return;
  int m   = idx >> 18;              // which matrix (512*512 = 2^18)
  int rem = idx & (DM * DM - 1);
  int r   = rem >> 9;               // k (row of W)
  int c   = rem & (DM - 1);         // n (col of W)
  const float* W = (m == 0) ? Wq : (m == 1) ? Wk : (m == 2) ? Wv : Wo;
  _Float16*    T = (m == 0) ? wtq : (m == 1) ? wtk : (m == 2) ? wtv : wto;
  T[(size_t)c * DM + r] = (_Float16)W[rem];
}

// ---- Kernel 1: fused QKV projection ---------------------------------------
// grid (32, 8, 3), block 256 (8 waves). Wave = 16 rows x 64 cols, K-loop 512.
// Weight slice (64 cols x 512 k, f16 = 64KB) staged into LDS by TDM once per
// block; B tiles then come from LDS (kills 8x redundant global streaming).

__global__ __launch_bounds__(256)
void qkv_proj(const float* __restrict__ x,
              const float* __restrict__ bq, const float* __restrict__ bk,
              const float* __restrict__ bv,
              const _Float16* __restrict__ wtq, const _Float16* __restrict__ wtk,
              const _Float16* __restrict__ wtv,
              _Float16* __restrict__ qh, _Float16* __restrict__ kh,
              _Float16* __restrict__ vt) {
  extern __shared__ _Float16 wlds[];            // [64][512] f16 = 64KB
  int wave    = threadIdx.x >> 5;
  int rowBase = blockIdx.x * 128 + wave * 16;   // 0..4095
  int colBase = blockIdx.y * 64;                // 0..511
  int m       = blockIdx.z;                     // 0=q 1=k 2=v
  const _Float16* wt   = (m == 0) ? wtq : (m == 1) ? wtk : wtv;
  const float*    bias = (m == 0) ? bq  : (m == 1) ? bk  : bv;

  if (wave == 0) {
    tdm_copy_to_lds(wt + (size_t)colBase * DM,
                    (unsigned)(uintptr_t)(void*)wlds, 64 * DM * 2);
    __builtin_amdgcn_s_wait_tensorcnt(0);
  }
  __syncthreads();

  const float* xrow = x + (size_t)rowBase * DM;
  v8f acc[4] = {v8f{}, v8f{}, v8f{}, v8f{}};
  v16h a = load_a_f32(xrow, DM, 0);
  for (int kc = 0; kc < DM; kc += 32) {
    v16h an = a;
    if (kc + 32 < DM) an = load_a_f32(xrow, DM, kc + 32);  // pipeline A
    // issue all 8 ds_load_b128 before the first WMMA -> staged dscnt waits
    v16h b0 = load_b_f16(wlds + (size_t)(0 * 16) * DM, DM, kc);
    v16h b1 = load_b_f16(wlds + (size_t)(1 * 16) * DM, DM, kc);
    v16h b2 = load_b_f16(wlds + (size_t)(2 * 16) * DM, DM, kc);
    v16h b3 = load_b_f16(wlds + (size_t)(3 * 16) * DM, DM, kc);
    acc[0] = wmma_f16(a, b0, acc[0]);
    acc[1] = wmma_f16(a, b1, acc[1]);
    acc[2] = wmma_f16(a, b2, acc[2]);
    acc[3] = wmma_f16(a, b3, acc[3]);
    a = an;
  }

  int lane = threadIdx.x & 31;
  int n    = lane & 15;
  int mb   = (lane >> 4) * 8;
  int bidx = rowBase >> 11;           // batch
  int s0   = rowBase & (SEQ - 1);     // seq position of row 0
#pragma unroll
  for (int t = 0; t < 4; ++t) {
    int   col = colBase + t * 16 + n;
    float bb  = bias[col];
    int   hh  = col >> 6;
    int   dd  = col & (DH - 1);
    if (m < 2) {
      _Float16* dst = ((m == 0) ? qh : kh) + (size_t)(bidx * NH + hh) * SEQ * DH;
#pragma unroll
      for (int g = 0; g < 8; ++g)
        dst[(size_t)(s0 + mb + g) * DH + dd] = (_Float16)(acc[t][g] + bb);
    } else {
      v4h pk0, pk1;
#pragma unroll
      for (int g = 0; g < 4; ++g) {
        pk0[g] = (_Float16)(acc[t][g] + bb);
        pk1[g] = (_Float16)(acc[t][4 + g] + bb);
      }
      _Float16* dst = vt + ((size_t)(bidx * NH + hh) * DH + dd) * SEQ + s0 + mb;
      *(v4h*)(dst)     = pk0;
      *(v4h*)(dst + 4) = pk1;
    }
  }
}

// ---- Kernel 2: attention (scores -> softmax -> P@V), one (b,h,16-q) tile ---
// block 256 (8 waves); dynamic LDS = 16*2048 fp32 scores + 8*1024 fp32 partials

__global__ __launch_bounds__(256)
void attention(const _Float16* __restrict__ qh, const _Float16* __restrict__ kh,
               const _Float16* __restrict__ vt,
               float* __restrict__ attn, _Float16* __restrict__ ctx) {
  extern __shared__ float smem[];
  float* probs   = smem;               // [16][2048]
  float* partial = smem + 16 * SEQ;    // [8][16*64]

  int bh    = blockIdx.x;
  int b     = bh >> 3;
  int h     = bh & 7;
  int qBase = blockIdx.y * 16;
  int wave  = threadIdx.x >> 5;
  int lane  = threadIdx.x & 31;
  int n     = lane & 15;
  int mb    = (lane >> 4) * 8;

  const _Float16* qptr = qh + ((size_t)bh * SEQ + qBase) * DH;
  const _Float16* kptr = kh + (size_t)bh * SEQ * DH;
  const _Float16* vptr = vt + (size_t)bh * DH * SEQ;

  // ---- Phase A: S = (Q K^T) * 1/sqrt(64); wave w owns keys [256w, 256w+256)
  v16h a0 = load_a_f16(qptr, DH, 0);
  v16h a1 = load_a_f16(qptr, DH, 32);
  int  kb0 = wave * 256;
  v16h b0 = load_b_f16(kptr + (size_t)kb0 * DH, DH, 0);
  v16h b1 = load_b_f16(kptr + (size_t)kb0 * DH, DH, 32);
#pragma unroll 4
  for (int t = 0; t < 16; ++t) {
    int  keyBase = wave * 256 + t * 16;
    v16h nb0 = b0, nb1 = b1;
    if (t < 15) {                      // software pipeline: next tile's loads
      nb0 = load_b_f16(kptr + (size_t)(keyBase + 16) * DH, DH, 0);
      nb1 = load_b_f16(kptr + (size_t)(keyBase + 16) * DH, DH, 32);
    }
    v8f s = {};
    s = wmma_f16(a0, b0, s);
    s = wmma_f16(a1, b1, s);
#pragma unroll
    for (int g = 0; g < 8; ++g)
      probs[(size_t)(mb + g) * SEQ + keyBase + n] = s[g] * 0.125f;
    b0 = nb0; b1 = nb1;
  }

  // warm L2 for the V slice this wave consumes in Phase C (global_prefetch_b8)
#pragma unroll
  for (int t = 0; t < 4; ++t)
    __builtin_prefetch(vptr + (size_t)(t * 16 + n) * SEQ + wave * 256 + mb * 16,
                       0, 2);
  __syncthreads();

  // ---- Phase B: softmax over each 2048-wide row; wave w owns rows 2w, 2w+1
  float* attnBase = attn + ((size_t)bh * SEQ + qBase) * SEQ;
  for (int rr = 0; rr < 2; ++rr) {
    int    r   = wave * 2 + rr;
    float* row = probs + (size_t)r * SEQ;
    float  mx  = -3.0e38f;
    for (int j = lane * 4; j < SEQ; j += 128) {
      v4f v = *(const v4f*)(row + j);
      mx = fmaxf(mx, fmaxf(fmaxf(v[0], v[1]), fmaxf(v[2], v[3])));
    }
#pragma unroll
    for (int off = 16; off > 0; off >>= 1) mx = fmaxf(mx, __shfl_xor(mx, off, 32));

    float sum = 0.0f;
    for (int j = lane * 4; j < SEQ; j += 128) {
      v4f v = *(const v4f*)(row + j);
      v4f e;
#pragma unroll
      for (int i = 0; i < 4; ++i) { e[i] = __expf(v[i] - mx); sum += e[i]; }
      *(v4f*)(row + j) = e;
    }
#pragma unroll
    for (int off = 16; off > 0; off >>= 1) sum += __shfl_xor(sum, off, 32);

    float  inv  = 1.0f / sum;
    float* arow = attnBase + (size_t)r * SEQ;
    for (int j = lane * 4; j < SEQ; j += 128) {
      v4f v = *(const v4f*)(row + j);
#pragma unroll
      for (int i = 0; i < 4; ++i) v[i] *= inv;
      *(v4f*)(row + j)  = v;   // keep normalized P in LDS for P@V
      *(v4f*)(arow + j) = v;   // attention-probability output (fp32, coalesced)
    }
  }
  __syncthreads();

  // ---- Phase C: ctx = P @ V ; k-split across 8 waves, LDS reduce.
  // All 4 B tiles (global) + next A tile (LDS) issued before the 4 WMMAs.
  v8f acc[4] = {v8f{}, v8f{}, v8f{}, v8f{}};
  v16h a = load_a_f32(probs, SEQ, wave * 256);
  for (int cc = 0; cc < 8; ++cc) {
    int  keyc = wave * 256 + cc * 32;
    v16h vb0 = load_b_f16(vptr + (size_t)(0 * 16) * SEQ, SEQ, keyc);
    v16h vb1 = load_b_f16(vptr + (size_t)(1 * 16) * SEQ, SEQ, keyc);
    v16h vb2 = load_b_f16(vptr + (size_t)(2 * 16) * SEQ, SEQ, keyc);
    v16h vb3 = load_b_f16(vptr + (size_t)(3 * 16) * SEQ, SEQ, keyc);
    v16h an = a;
    if (cc < 7) an = load_a_f32(probs, SEQ, keyc + 32);   // pipeline A
    acc[0] = wmma_f16(a, vb0, acc[0]);
    acc[1] = wmma_f16(a, vb1, acc[1]);
    acc[2] = wmma_f16(a, vb2, acc[2]);
    acc[3] = wmma_f16(a, vb3, acc[3]);
    a = an;
  }
#pragma unroll
  for (int t = 0; t < 4; ++t)
#pragma unroll
    for (int g = 0; g < 8; ++g)
      partial[wave * 1024 + (mb + g) * 64 + t * 16 + n] = acc[t][g];
  __syncthreads();

  int base = threadIdx.x * 4;                 // 0..1020, covers 16x64
  v4f sum4 = {};
#pragma unroll
  for (int w = 0; w < 8; ++w) {
    v4f p = *(const v4f*)(partial + w * 1024 + base);
    sum4 += p;
  }
  int mrow = base >> 6;
  int dd   = base & (DH - 1);
  v4h pk;
#pragma unroll
  for (int i = 0; i < 4; ++i) pk[i] = (_Float16)sum4[i];
  _Float16* cdst = ctx + ((size_t)b * SEQ + qBase + mrow) * DM + h * DH + dd;
  *(v4h*)cdst = pk;
}

// ---- Kernel 3: out = ctx @ Wo + bo (Wo slice staged to LDS via TDM) --------

__global__ __launch_bounds__(256)
void out_proj(const _Float16* __restrict__ ctx, const _Float16* __restrict__ wto,
              const float* __restrict__ bo, float* __restrict__ out) {
  extern __shared__ _Float16 olds[];            // [64][512] f16 = 64KB
  int wave    = threadIdx.x >> 5;
  int rowBase = blockIdx.x * 128 + wave * 16;
  int colBase = blockIdx.y * 64;

  if (wave == 0) {
    tdm_copy_to_lds(wto + (size_t)colBase * DM,
                    (unsigned)(uintptr_t)(void*)olds, 64 * DM * 2);
    __builtin_amdgcn_s_wait_tensorcnt(0);
  }
  __syncthreads();

  const _Float16* crow = ctx + (size_t)rowBase * DM;
  v8f acc[4] = {v8f{}, v8f{}, v8f{}, v8f{}};
  v16h a = load_a_f16(crow, DM, 0);
  for (int kc = 0; kc < DM; kc += 32) {
    v16h an = a;
    if (kc + 32 < DM) an = load_a_f16(crow, DM, kc + 32);   // pipeline A
    v16h b0 = load_b_f16(olds + (size_t)(0 * 16) * DM, DM, kc);
    v16h b1 = load_b_f16(olds + (size_t)(1 * 16) * DM, DM, kc);
    v16h b2 = load_b_f16(olds + (size_t)(2 * 16) * DM, DM, kc);
    v16h b3 = load_b_f16(olds + (size_t)(3 * 16) * DM, DM, kc);
    acc[0] = wmma_f16(a, b0, acc[0]);
    acc[1] = wmma_f16(a, b1, acc[1]);
    acc[2] = wmma_f16(a, b2, acc[2]);
    acc[3] = wmma_f16(a, b3, acc[3]);
    a = an;
  }
  int lane = threadIdx.x & 31;
  int n    = lane & 15;
  int mb   = (lane >> 4) * 8;
#pragma unroll
  for (int t = 0; t < 4; ++t) {
    int   col = colBase + t * 16 + n;
    float bb  = bo[col];
#pragma unroll
    for (int g = 0; g < 8; ++g)
      out[(size_t)(rowBase + mb + g) * DM + col] = acc[t][g] + bb;
  }
}

// ---- Host orchestration ----------------------------------------------------

extern "C" void kernel_launch(void* const* d_in, const int* in_sizes, int n_in,
                              void* d_out, int out_size, void* d_ws, size_t ws_size,
                              hipStream_t stream) {
  (void)in_sizes; (void)n_in; (void)out_size; (void)ws_size;
  const float* x  = (const float*)d_in[0];
  const float* Wq = (const float*)d_in[1];
  const float* bq = (const float*)d_in[2];
  const float* Wk = (const float*)d_in[3];
  const float* bk = (const float*)d_in[4];
  const float* Wv = (const float*)d_in[5];
  const float* bv = (const float*)d_in[6];
  const float* Wo = (const float*)d_in[7];
  const float* bo = (const float*)d_in[8];

  float* out  = (float*)d_out;
  float* attn = out + (size_t)BATCH * SEQ * DM;   // outputs concatenated: (out, attn)

  char* ws = (char*)d_ws;
  const size_t wBytes = (size_t)DM * DM * sizeof(_Float16);
  const size_t hBytes = (size_t)BATCH * NH * SEQ * DH * sizeof(_Float16);
  _Float16* wtq = (_Float16*)ws;                 ws += wBytes;
  _Float16* wtk = (_Float16*)ws;                 ws += wBytes;
  _Float16* wtv = (_Float16*)ws;                 ws += wBytes;
  _Float16* wto = (_Float16*)ws;                 ws += wBytes;
  _Float16* qh  = (_Float16*)ws;                 ws += hBytes;
  _Float16* kh  = (_Float16*)ws;                 ws += hBytes;
  _Float16* vt  = (_Float16*)ws;                 ws += hBytes;
  _Float16* ctx = (_Float16*)ws;                 ws += hBytes;

  const size_t projSmem = (size_t)64 * DM * sizeof(_Float16);              // 64 KB
  const size_t attnSmem = (size_t)(16 * SEQ + 8 * 1024) * sizeof(float);   // 160 KB
  static bool attrSet = false;   // idempotent host-side attributes, not stream work
  if (!attrSet) {
    (void)hipFuncSetAttribute(reinterpret_cast<const void*>(qkv_proj),
                              hipFuncAttributeMaxDynamicSharedMemorySize,
                              (int)projSmem);
    (void)hipFuncSetAttribute(reinterpret_cast<const void*>(attention),
                              hipFuncAttributeMaxDynamicSharedMemorySize,
                              (int)attnSmem);
    (void)hipFuncSetAttribute(reinterpret_cast<const void*>(out_proj),
                              hipFuncAttributeMaxDynamicSharedMemorySize,
                              (int)projSmem);
    attrSet = true;
  }

  prep_weights<<<(4 * DM * DM + 255) / 256, 256, 0, stream>>>(
      Wq, Wk, Wv, Wo, wtq, wtk, wtv, wto);

  qkv_proj<<<dim3(32, 8, 3), 256, projSmem, stream>>>(
      x, bq, bk, bv, wtq, wtk, wtv, qh, kh, vt);

  attention<<<dim3(BATCH * NH, SEQ / 16), 256, attnSmem, stream>>>(
      qh, kh, vt, attn, ctx);

  out_proj<<<dim3(32, 8), 256, projSmem, stream>>>(ctx, wto, bo, out);
}
